// ExpertChoiceFFN_72438918414450
// MI455X (gfx1250) — matrix operand order repaired and checked by reading, not compiled, not run on hardware
//
#include <hip/hip_runtime.h>
#include <hip/hip_bf16.h>

// ---------------------------------------------------------------------------
// ExpertChoiceFFN for MI455X (gfx1250): split-bf16 WMMA grouped GEMM pipeline
//   b=4, s=2048, h=2048 -> bs=8192 tokens; E=16 experts; k=512 tokens/expert
// Round 3: TDM (tensor_load_to_lds, 6-arg clang-23 form) B staging from
// preconverted bf16 W with hardware LDS padding; fallback if ws too small.
// ---------------------------------------------------------------------------

#define HID   2048
#define BS    8192
#define NEXP  16
#define KTOP  512
#define LDSW  40      // padded row stride in bf16 elements (80B, 16B aligned)

typedef __attribute__((ext_vector_type(16))) __bf16 v16bf;
typedef __attribute__((ext_vector_type(8)))  float  v8f;
typedef __attribute__((ext_vector_type(4)))  unsigned int u32x4;
typedef __attribute__((ext_vector_type(8)))  int i32x8;
typedef __attribute__((ext_vector_type(4)))  int i32x4;

struct U4 { unsigned int x, y, z, w; } __attribute__((aligned(16)));
struct B32B { U4 lo, hi; };  // 32 bytes == sizeof(v16bf)

__device__ __forceinline__ unsigned short bf16_rne(float f) {
  unsigned int u = __float_as_uint(f);
  u += 0x7FFFu + ((u >> 16) & 1u);
  return (unsigned short)(u >> 16);
}

// split f into hi (bf16 RNE) and lo (bf16 of residual); hi+lo ~ f to ~2^-16 rel
__device__ __forceinline__ void split2(float f, unsigned short& h, unsigned short& l) {
  h = bf16_rne(f);
  float fh = __uint_as_float(((unsigned int)h) << 16);
  l = bf16_rne(f - fh);
}

// Build a v16bf fragment from two 16B LDS chunks (p0: 8 bf16, p1: 8 bf16)
__device__ __forceinline__ v16bf ld_frag(const unsigned short* p0,
                                         const unsigned short* p1) {
  B32B t;
  t.lo = *(const U4*)p0;
  t.hi = *(const U4*)p1;
  return __builtin_bit_cast(v16bf, t);
}

__device__ __forceinline__ v8f wmma_bf16(v16bf a, v16bf b, v8f c) {
  return __builtin_amdgcn_wmma_f32_16x16x32_bf16(false, a, false, b,
                                                 (short)0, c, false, false);
}

// Convert 16 consecutive fp32 (4x float4) into hi/lo bf16, store 2x16B each.
__device__ __forceinline__ void cvt_row16(const float* src,
                                          unsigned short* hi_dst,
                                          unsigned short* lo_dst) {
  const float4* p = (const float4*)src;
  float4 a = p[0], b = p[1], c = p[2], d = p[3];
  float v[16] = {a.x, a.y, a.z, a.w, b.x, b.y, b.z, b.w,
                 c.x, c.y, c.z, c.w, d.x, d.y, d.z, d.w};
  unsigned int hp[8], lp[8];
#pragma unroll
  for (int j = 0; j < 8; ++j) {
    unsigned short h0, l0, h1, l1;
    split2(v[2*j],   h0, l0);
    split2(v[2*j+1], h1, l1);
    hp[j] = (unsigned int)h0 | ((unsigned int)h1 << 16);
    lp[j] = (unsigned int)l0 | ((unsigned int)l1 << 16);
  }
  ((U4*)hi_dst)[0] = U4{hp[0], hp[1], hp[2], hp[3]};
  ((U4*)hi_dst)[1] = U4{hp[4], hp[5], hp[6], hp[7]};
  ((U4*)lo_dst)[0] = U4{lp[0], lp[1], lp[2], lp[3]};
  ((U4*)lo_dst)[1] = U4{lp[4], lp[5], lp[6], lp[7]};
}

// LDS byte offset of a __shared__ object: generic LDS pointers carry the LDS
// offset in the low 32 bits (ISA 10.2 aperture mapping).
__device__ __forceinline__ unsigned lds_byte_off(const void* p) {
  return (unsigned)(unsigned long long)p;
}

// Issue a TDM 2D tile load: tile_d1 rows x tile_d0 elements (2B each) from a
// row-major tensor with row stride `stride_elems`, into LDS at lds_off with
// a 4-DWORD pad inserted after every 16 DWORDs (=> 80B LDS row stride).
__device__ __forceinline__ void tdm_load_2d(unsigned lds_off, const void* gptr,
                                            unsigned tile_d0, unsigned tile_d1,
                                            unsigned stride_elems) {
  unsigned long long ga = (unsigned long long)gptr;
  u32x4 g0;
  g0[0] = 0x1u;                                   // count=1, user descriptor
  g0[1] = lds_off;                                // lds_addr (bytes)
  g0[2] = (unsigned)ga;                           // global_addr[31:0]
  g0[3] = (unsigned)((ga >> 32) & 0x1FFFFFFu)     // global_addr[56:32]
        | (2u << 30);                             // type=2 ("image")
  i32x8 g1;
  g1[0] = (int)((1u << 16)      // data_size = 1 -> 2 bytes
              | (1u << 20)      // pad_enable
              | (3u << 22)      // pad_interval: 16 DWORDs (64B) between pads
              | (3u << 25));    // pad_amount: 4 DWORDs (16B)
  g1[1] = (int)(((unsigned)HID & 0xFFFFu) << 16); // tensor_dim0[15:0]
  g1[2] = (int)((((unsigned)HID >> 16) & 0xFFFFu)
              | (((unsigned)HID & 0xFFFFu) << 16));  // tdim0 hi | tdim1 lo
  g1[3] = (int)((((unsigned)HID >> 16) & 0xFFFFu)
              | (tile_d0 << 16));                 // tdim1 hi | tile_dim0
  g1[4] = (int)(tile_d1 & 0xFFFFu);               // tile_dim1 | tile_dim2=0
  g1[5] = (int)stride_elems;                      // tensor_dim0_stride[31:0]
  g1[6] = 0;                                      // stride0 hi | stride1 lo
  g1[7] = 0;
  i32x4 z4 = {0, 0, 0, 0};
  i32x8 z8 = {0, 0, 0, 0, 0, 0, 0, 0};
  __builtin_amdgcn_tensor_load_to_lds(g0, g1, z4, z4, z8, 0);
}

// ---------------------------------------------------------------------------
// Kernel 0: zero the output (harness poisons d_out; scatter-add needs zeros)
// ---------------------------------------------------------------------------
__global__ void zero_f4(float4* __restrict__ p, int n4) {
  int i = blockIdx.x * blockDim.x + threadIdx.x;
  if (i < n4) p[i] = make_float4(0.f, 0.f, 0.f, 0.f);
}

// ---------------------------------------------------------------------------
// Kernel 1: transpose + split expert W (fp32 [k][n]) -> Wt_hi/Wt_lo (bf16
// [n][k]). One pass; removes 64x redundant conversion from the GEMM loop and
// gives TDM a K-contiguous source. Grid: (32,32) x 256 threads, 64x64 tiles.
// ---------------------------------------------------------------------------
__global__ __launch_bounds__(256)
void convert_w_t(const float* __restrict__ W, unsigned short* __restrict__ Wh,
                 unsigned short* __restrict__ Wl) {
  __shared__ float tile[64][65];
  const int n0 = blockIdx.x * 64, k0 = blockIdx.y * 64;
  for (int i = threadIdx.x; i < 64 * 64; i += 256) {
    int r = i >> 6, c = i & 63;                    // c contiguous over n
    tile[r][c] = W[(size_t)(k0 + r) * HID + n0 + c];
  }
  __syncthreads();
  for (int i = threadIdx.x; i < 64 * 64; i += 256) {
    int r = i >> 6, c = i & 63;                    // c contiguous over k
    unsigned short h, l;
    split2(tile[c][r], h, l);
    Wh[(size_t)(n0 + r) * HID + k0 + c] = h;
    Wl[(size_t)(n0 + r) * HID + k0 + c] = l;
  }
}

// ---------------------------------------------------------------------------
// Kernel 2: router GEMM (split-bf16 WMMA, N=16 == one tile) + fused softmax.
// Writes ST[e][t] = softmax prob, transposed for coalesced top-k reads.
// ---------------------------------------------------------------------------
__global__ __launch_bounds__(256)
void router_softmax(const float* __restrict__ x, const float* __restrict__ rw,
                    const float* __restrict__ rb, float* __restrict__ ST) {
  __shared__ unsigned short As_hi[128 * LDSW];
  __shared__ unsigned short As_lo[128 * LDSW];
  __shared__ unsigned short Bt_hi[16 * LDSW];
  __shared__ unsigned short Bt_lo[16 * LDSW];
  __shared__ float lg[128 * 16];

  const int tid  = threadIdx.x;
  const int m0   = blockIdx.x * 128;
  const int lane = tid & 31;
  const int wave = tid >> 5;          // 8 waves, each owns a 16-row tile
  const int kh   = lane >> 4;
  const int ml   = lane & 15;

  const int ar  = tid >> 1;           // A-fill: row within tile
  const int ah2 = tid & 1;            // A-fill: k half
  const float* arow = x + (size_t)(m0 + ar) * HID;

  v8f acc = {};

  for (int k0 = 0; k0 < HID; k0 += 32) {
    cvt_row16(arow + k0 + ah2 * 16,
              &As_hi[ar * LDSW + ah2 * 16], &As_lo[ar * LDSW + ah2 * 16]);
#pragma unroll
    for (int q = 0; q < 2; ++q) {
      int el = tid * 2 + q;           // 512 elements over 256 threads
      int bk = el >> 4, bn = el & 15;
      unsigned short h, l;
      split2(rw[(size_t)(k0 + bk) * NEXP + bn], h, l);
      Bt_hi[bn * LDSW + bk] = h;
      Bt_lo[bn * LDSW + bk] = l;
    }
    __syncthreads();

    const unsigned short* pah = &As_hi[(wave * 16 + ml) * LDSW + kh * 8];
    const unsigned short* pal = &As_lo[(wave * 16 + ml) * LDSW + kh * 8];
    const unsigned short* pbh = &Bt_hi[ml * LDSW + kh * 8];
    const unsigned short* pbl = &Bt_lo[ml * LDSW + kh * 8];
    v16bf a_h = ld_frag(pah, pah + 16);
    v16bf a_l = ld_frag(pal, pal + 16);
    v16bf b_h = ld_frag(pbh, pbh + 16);
    v16bf b_l = ld_frag(pbl, pbl + 16);

    acc = wmma_bf16(a_h, b_h, acc);   // hi*hi
    acc = wmma_bf16(a_h, b_l, acc);   // hi*lo
    acc = wmma_bf16(a_l, b_h, acc);   // lo*hi  (lo*lo negligible)
    __syncthreads();
  }

  float rbn = rb[ml];
#pragma unroll
  for (int g = 0; g < 8; ++g)
    lg[(wave * 16 + g + 8 * kh) * 16 + ml] = acc[g] + rbn;
  __syncthreads();

  if (tid < 128) {
    float v[16], mx = -1e30f;
#pragma unroll
    for (int i = 0; i < 16; ++i) { v[i] = lg[tid * 16 + i]; mx = fmaxf(mx, v[i]); }
    float s = 0.f;
#pragma unroll
    for (int i = 0; i < 16; ++i) { v[i] = __expf(v[i] - mx); s += v[i]; }
    float inv = 1.f / s;
#pragma unroll
    for (int i = 0; i < 16; ++i)
      ST[(size_t)i * BS + m0 + tid] = v[i] * inv;
  }
}

// ---------------------------------------------------------------------------
// Kernel 3: per-expert top-512 via in-LDS bitonic sort of 8192 (score, idx).
// Grid: 16 blocks (one per expert) x 1024 threads. 48KB LDS.
// ---------------------------------------------------------------------------
__global__ __launch_bounds__(1024)
void topk_expert(const float* __restrict__ ST, int* __restrict__ tki,
                 float* __restrict__ tkg) {
  __shared__ float          key[BS];
  __shared__ unsigned short val[BS];
  const int e = blockIdx.x, tid = threadIdx.x;

  for (int i = tid; i < BS; i += 1024) {
    key[i] = ST[(size_t)e * BS + i];
    val[i] = (unsigned short)i;
  }
  __syncthreads();

  for (int k = 2; k <= BS; k <<= 1) {
    for (int j = k >> 1; j > 0; j >>= 1) {
      for (int i = tid; i < BS; i += 1024) {
        int l = i ^ j;
        if (l > i) {
          bool desc = ((i & k) == 0);            // overall descending sort
          float ki = key[i], kl = key[l];
          if (desc ? (ki < kl) : (ki > kl)) {
            key[i] = kl; key[l] = ki;
            unsigned short t = val[i]; val[i] = val[l]; val[l] = t;
          }
        }
      }
      __syncthreads();
    }
  }

  for (int r = tid; r < KTOP; r += 1024) {
    tki[e * KTOP + r] = (int)val[r];
    tkg[e * KTOP + r] = key[r];
  }
}

// ---------------------------------------------------------------------------
// Kernel 4: grouped GEMM (split-bf16 WMMA) + gate weighting + atomic scatter.
// USE_TDM=true: B tiles staged by the Tensor Data Mover from preconverted
// bf16 Wt (n-major), with hardware pad -> 80B LDS row stride; completion via
// TENSORcnt. USE_TDM=false: round-1 in-kernel fp32->bf16 conversion.
// Grid: (16 n-blocks, 4 m-blocks, 16 experts) x 256 threads.
// ---------------------------------------------------------------------------
template <bool USE_TDM>
__global__ __launch_bounds__(256)
void moe_gemm_scatter(const float* __restrict__ x, const float* __restrict__ W,
                      const unsigned short* __restrict__ Wth,
                      const unsigned short* __restrict__ Wtl,
                      const float* __restrict__ eb, const int* __restrict__ tki,
                      const float* __restrict__ tkg, float* __restrict__ y) {
  __shared__ unsigned short As_hi[128 * LDSW];
  __shared__ unsigned short As_lo[128 * LDSW];
  __shared__ unsigned short Bt_hi[128 * LDSW];
  __shared__ unsigned short Bt_lo[128 * LDSW];
  __shared__ int   idx_l[128];
  __shared__ float g_l[128];

  const int tid  = threadIdx.x;
  const int n0g  = blockIdx.x * 128;  // output column base
  const int m0s  = blockIdx.y * 128;  // slot base within expert
  const int e    = blockIdx.z;

  if (tid < 128) {
    idx_l[tid] = tki[e * KTOP + m0s + tid];
    g_l[tid]   = tkg[e * KTOP + m0s + tid];
  }
  __syncthreads();

  const int lane = tid & 31;
  const int wave = tid >> 5;
  const int wm   = wave & 3;          // row group: 32*wm
  const int wn   = wave >> 2;         // col group: 64*wn
  const int kh   = lane >> 4;
  const int ml   = lane & 15;

  const int ar  = tid >> 1;           // A-fill row
  const int ah2 = tid & 1;            // A-fill k half
  const float* arow = x + (size_t)idx_l[ar] * HID;
  const int bk = tid >> 3;            // B-fill K row (fallback path)
  const int bq = tid & 7;             // B-fill 16-col chunk (fallback path)

  const unsigned ldsBh = lds_byte_off(Bt_hi);
  const unsigned ldsBl = lds_byte_off(Bt_lo);

  v8f acc[2][4];
#pragma unroll
  for (int i = 0; i < 2; ++i)
#pragma unroll
    for (int j = 0; j < 4; ++j) acc[i][j] = (v8f){};

  for (int k0 = 0; k0 < HID; k0 += 32) {
    if constexpr (USE_TDM) {
      // B tiles: 128 n-rows x 32 k-elements of bf16 Wt, DMA'd by the TDM.
      if (wave == 0) {
        tdm_load_2d(ldsBh, Wth + (size_t)n0g * HID + k0, 32u, 128u, HID);
        tdm_load_2d(ldsBl, Wtl + (size_t)n0g * HID + k0, 32u, 128u, HID);
      }
    } else {
      const float4* bp =
          (const float4*)(W + (size_t)(k0 + bk) * HID + n0g + bq * 16);
      float4 f0 = bp[0], f1 = bp[1], f2 = bp[2], f3 = bp[3];
      float bv[16] = {f0.x, f0.y, f0.z, f0.w, f1.x, f1.y, f1.z, f1.w,
                      f2.x, f2.y, f2.z, f2.w, f3.x, f3.y, f3.z, f3.w};
#pragma unroll
      for (int j = 0; j < 16; ++j) {
        unsigned short h, l;
        split2(bv[j], h, l);
        int nl = bq * 16 + j;
        Bt_hi[nl * LDSW + bk] = h;
        Bt_lo[nl * LDSW + bk] = l;
      }
    }
    // A tile: gathered token rows fp32 -> hi/lo bf16 (overlaps the TDM DMA)
    cvt_row16(arow + k0 + ah2 * 16,
              &As_hi[ar * LDSW + ah2 * 16], &As_lo[ar * LDSW + ah2 * 16]);
    __builtin_prefetch(arow + k0 + 32, 0, 1);   // global_prefetch next A tile
    if constexpr (USE_TDM) {
      if (wave == 0) __builtin_amdgcn_s_wait_tensorcnt(0);
    }
    __syncthreads();

    v16bf a_h[2], a_l[2], b_h[4], b_l[4];
#pragma unroll
    for (int mt = 0; mt < 2; ++mt) {
      const unsigned short* ph = &As_hi[(wm*32 + mt*16 + ml) * LDSW + kh * 8];
      const unsigned short* pl = &As_lo[(wm*32 + mt*16 + ml) * LDSW + kh * 8];
      a_h[mt] = ld_frag(ph, ph + 16);
      a_l[mt] = ld_frag(pl, pl + 16);
    }
#pragma unroll
    for (int nt = 0; nt < 4; ++nt) {
      const unsigned short* ph = &Bt_hi[(wn*64 + nt*16 + ml) * LDSW + kh * 8];
      const unsigned short* pl = &Bt_lo[(wn*64 + nt*16 + ml) * LDSW + kh * 8];
      b_h[nt] = ld_frag(ph, ph + 16);
      b_l[nt] = ld_frag(pl, pl + 16);
    }
#pragma unroll
    for (int mt = 0; mt < 2; ++mt)
#pragma unroll
      for (int nt = 0; nt < 4; ++nt) {
        acc[mt][nt] = wmma_bf16(a_h[mt], b_h[nt], acc[mt][nt]);  // hi*hi
        acc[mt][nt] = wmma_bf16(a_h[mt], b_l[nt], acc[mt][nt]);  // hi*lo
        acc[mt][nt] = wmma_bf16(a_l[mt], b_h[nt], acc[mt][nt]);  // lo*hi
      }
    __syncthreads();
  }

  // gate-weight + bias, scatter-add to token rows (native f32 atomics in L2)
#pragma unroll
  for (int nt = 0; nt < 4; ++nt) {
    const int colg = n0g + wn * 64 + nt * 16 + ml;
    const float ebn = eb[colg];
#pragma unroll
    for (int mt = 0; mt < 2; ++mt) {
      v8f c = acc[mt][nt];
#pragma unroll
      for (int g = 0; g < 8; ++g) {
        int slot = wm * 32 + mt * 16 + 8 * kh + g;   // C layout: M = g + 8*kh
        float v = g_l[slot] * (c[g] + ebn);
        float* dst = &y[(size_t)idx_l[slot] * HID + colg];
        __hip_atomic_fetch_add(dst, v, __ATOMIC_RELAXED,
                               __HIP_MEMORY_SCOPE_AGENT);
      }
    }
  }
}

// ---------------------------------------------------------------------------
extern "C" void kernel_launch(void* const* d_in, const int* in_sizes, int n_in,
                              void* d_out, int out_size, void* d_ws,
                              size_t ws_size, hipStream_t stream) {
  const float* x  = (const float*)d_in[0];   // (4,2048,2048)
  const float* rw = (const float*)d_in[1];   // (2048,16)
  const float* rb = (const float*)d_in[2];   // (16,)
  const float* ew = (const float*)d_in[3];   // (2048,2048)
  const float* eb = (const float*)d_in[4];   // (2048,)
  float* y = (float*)d_out;                  // (4,2048,2048)

  // workspace: ST | topk_idx | topk_g | [Wt_hi | Wt_lo at 1MB if room]
  float* ST  = (float*)d_ws;
  int*   tki = (int*)((char*)d_ws + (size_t)NEXP * BS * sizeof(float));
  float* tkg = (float*)((char*)tki + (size_t)NEXP * KTOP * sizeof(int));
  unsigned short* Wth = (unsigned short*)((char*)d_ws + (1u << 20));
  unsigned short* Wtl = Wth + (size_t)HID * HID;
  const size_t need_tdm = (1u << 20) + 2 * (size_t)HID * HID * sizeof(unsigned short);
  const bool use_tdm = (ws_size >= need_tdm);

  int n4 = out_size / 4;
  zero_f4<<<(n4 + 255) / 256, 256, 0, stream>>>((float4*)y, n4);
  router_softmax<<<BS / 128, 256, 0, stream>>>(x, rw, rb, ST);
  topk_expert<<<NEXP, 1024, 0, stream>>>(ST, tki, tkg);
  if (use_tdm) {
    convert_w_t<<<dim3(HID / 64, HID / 64), 256, 0, stream>>>(ew, Wth, Wtl);
    moe_gemm_scatter<true><<<dim3(HID / 128, KTOP / 128, NEXP), 256, 0,
                             stream>>>(x, ew, Wth, Wtl, eb, tki, tkg, y);
  } else {
    moe_gemm_scatter<false><<<dim3(HID / 128, KTOP / 128, NEXP), 256, 0,
                              stream>>>(x, ew, Wth, Wtl, eb, tki, tkg, y);
  }
}
